// Attention_64347200029308
// MI455X (gfx1250) — matrix operand order repaired and checked by reading, not compiled, not run on hardware
//
#include <hip/hip_runtime.h>

// ---------------------------------------------------------------------------
// Fused MHA forward for gfx1250 (MI455X), bf16 WMMA + fp32 accumulate.
// d_out = [ out: B*S*E fp32 | weights: B*H*S*S fp32 ] (reference returns both).
// Workspace: qh[B,H,S,DH] kh[B,H,S,DH] vT[B,H,DH,S] attnX[B,S,E], all bf16 (32MB).
// Roofline: the 512MB weights write dominates (~22us @ 23.3TB/s); all matrix
// math is bf16 WMMA so compute stays far below the memory floor.
// ---------------------------------------------------------------------------

typedef __attribute__((ext_vector_type(16))) __bf16 v16bf;
typedef __attribute__((ext_vector_type(8)))  __bf16 v8bf;
typedef __attribute__((ext_vector_type(8)))  float  v8f;
typedef __attribute__((ext_vector_type(4)))  float  v4f;

static constexpr int Bb = 2, Ss = 2048, Ee = 1024, Hh = 16, DHd = 64;

__device__ __forceinline__ v8f wmma_bf16(v16bf a, v16bf b, v8f c) {
  // D = A(16x32 bf16) x B(32x16 bf16) + C(16x16 f32)
  return __builtin_amdgcn_wmma_f32_16x16x32_bf16(
      /*neg_a=*/false, a, /*neg_b=*/false, b,
      /*c_mod=*/(short)0, c, /*reuse_a=*/false, /*reuse_b=*/false);
}

// A-fragment (16x32, bf16): lane<16 -> row=lane, k=k0..k0+15
//                           lane>=16 -> row=lane-16, k=k0+16..k0+31.
__device__ __forceinline__ v16bf load_a_bf16(const __bf16* __restrict__ base, int ld,
                                             int row0, int k0, int lane) {
  const __bf16* p = base + (size_t)(row0 + (lane & 15)) * ld + k0 + ((lane & 16) ? 16 : 0);
  v8bf lo = *(const v8bf*)p;
  v8bf hi = *(const v8bf*)(p + 8);
  return __builtin_shufflevector(lo, hi, 0,1,2,3,4,5,6,7,8,9,10,11,12,13,14,15);
}

// B-fragment (32x16, bf16): lane holds column n = col0+(lane&15);
// lanes 0-15 hold K=k0..k0+15, lanes 16-31 hold K=k0+16..k0+31 (contiguous 32B).
// Source is row-major [n, k] (i.e. K^T / V^T rows), so this is one vector load.
__device__ __forceinline__ v16bf load_b_bf16(const __bf16* __restrict__ base, int ld,
                                             int col0, int k0, int lane) {
  const __bf16* p = base + (size_t)(col0 + (lane & 15)) * ld + k0 + ((lane & 16) ? 16 : 0);
  return *(const v16bf*)p;
}

// Prefetch the line this lane's B-fragment will need later (global_prefetch_b8).
__device__ __forceinline__ void prefetch_b(const __bf16* __restrict__ base, int ld,
                                           int col0, int k0, int lane) {
  const __bf16* p = base + (size_t)(col0 + (lane & 15)) * ld + k0 + ((lane & 16) ? 16 : 0);
  __builtin_prefetch((const void*)p, 0, 1);
}

// Same fragments but sourced from fp32 with on-the-fly bf16 conversion.
__device__ __forceinline__ v16bf load_a_f32(const float* __restrict__ base, int ld,
                                            int row0, int k0, int lane) {
  const float* p = base + (size_t)(row0 + (lane & 15)) * ld + k0 + ((lane & 16) ? 16 : 0);
  v4f x0 = *(const v4f*)p, x1 = *(const v4f*)(p + 4);
  v4f x2 = *(const v4f*)(p + 8), x3 = *(const v4f*)(p + 12);
  v16bf f;
#pragma unroll
  for (int i = 0; i < 4; ++i) {
    f[i] = (__bf16)x0[i]; f[4 + i] = (__bf16)x1[i];
    f[8 + i] = (__bf16)x2[i]; f[12 + i] = (__bf16)x3[i];
  }
  return f;
}

__device__ __forceinline__ v16bf load_b_f32(const float* __restrict__ base, int ld,
                                            int col0, int k0, int lane) {
  const float* p = base + (size_t)(col0 + (lane & 15)) * ld + k0 + ((lane & 16) ? 16 : 0);
  v16bf f;
#pragma unroll
  for (int c = 0; c < 4; ++c) {
    v4f x = *(const v4f*)(p + 4 * c);
#pragma unroll
    for (int i = 0; i < 4; ++i) f[c * 4 + i] = (__bf16)x[i];
  }
  return f;
}

// ---------------------------------------------------------------------------
// Kernel 1: Xp = (X[4096,1024] @ W^T + bias) * scale -> bf16 head layout.
// transposed==0: out[b][h][s][dh] ; transposed==1: out[b][h][dh][s] (for V).
// Wave tile: 32(M) x 64(N), 8 WMMA accumulators, K-loop step 32.
// ---------------------------------------------------------------------------
__global__ void __launch_bounds__(256)
proj_kernel(const float* __restrict__ X, const float* __restrict__ W,
            const float* __restrict__ bias, __bf16* __restrict__ out,
            float scale, int transposed) {
  const int lane = threadIdx.x & 31;
  const int m0 = (blockIdx.y * 8 + (threadIdx.x >> 5)) * 32;
  const int n0 = blockIdx.x * 64;

  const v8f vzero = {0.f, 0.f, 0.f, 0.f, 0.f, 0.f, 0.f, 0.f};
  v8f acc[2][4];
#pragma unroll
  for (int mi = 0; mi < 2; ++mi)
#pragma unroll
    for (int j = 0; j < 4; ++j) acc[mi][j] = vzero;

  for (int k0 = 0; k0 < Ee; k0 += 32) {
    v16bf a0 = load_a_f32(X, Ee, m0, k0, lane);
    v16bf a1 = load_a_f32(X, Ee, m0 + 16, k0, lane);
    if (k0 + 128 < Ee) {  // prefetch the K-stream 4 steps ahead
      __builtin_prefetch((const void*)(X + (size_t)(m0 + (lane & 15)) * Ee + k0 + 128), 0, 1);
      __builtin_prefetch((const void*)(W + (size_t)(n0 + (lane & 15)) * Ee + k0 + 128), 0, 1);
    }
#pragma unroll
    for (int j = 0; j < 4; ++j) {
      v16bf bf = load_b_f32(W, Ee, n0 + j * 16, k0, lane);  // row n of W == col n of W^T
      acc[0][j] = wmma_bf16(a0, bf, acc[0][j]);
      acc[1][j] = wmma_bf16(a1, bf, acc[1][j]);
    }
  }

  // C layout: VGPR i, lanes 0-15 -> row m0+i, lanes 16-31 -> row m0+i+8; col = lane&15.
#pragma unroll
  for (int j = 0; j < 4; ++j) {
    const int n = n0 + j * 16 + (lane & 15);
    const float bn = bias[n];
    const int h = n >> 6, dh = n & 63;
#pragma unroll
    for (int mi = 0; mi < 2; ++mi)
#pragma unroll
      for (int i = 0; i < 8; ++i) {
        const int m = m0 + mi * 16 + i + ((lane & 16) ? 8 : 0);
        const int b = m >> 11, s = m & (Ss - 1);
        const float v = (acc[mi][j][i] + bn) * scale;
        const size_t idx = transposed
            ? ((size_t)((b * Hh + h) * DHd + dh)) * Ss + s
            : ((size_t)((b * Hh + h) * Ss + s)) * DHd + dh;
        out[idx] = (__bf16)v;
      }
  }
}

// ---------------------------------------------------------------------------
// Kernel 2: per (b,h): scores = qh @ kh^T (q pre-scaled), softmax -> weights
// (fp32, coalesced b128 stores), out = P @ V -> attnX bf16.
// One wave per 16-query block. Two-pass softmax:
//   pass 1: lane-LOCAL online max/sum (no shuffles in the loop), single
//           cross-lane reduction at the end; shift = m + ln(sum) folds the
//           normalization into the exponent.
//   pass 2: recompute scores, p = exp(x - shift); stage 16x32 P-tile in LDS
//           (stride 36 = conflict-free) to (a) emit coalesced 128B weight
//           stores and (b) reshape C-layout -> A-fragment for P@V WMMAs.
// ---------------------------------------------------------------------------
__global__ void __launch_bounds__(256)
attn_kernel(const __bf16* __restrict__ qh, const __bf16* __restrict__ kh,
            const __bf16* __restrict__ vT, float* __restrict__ wout,
            __bf16* __restrict__ attnX) {
  __shared__ __align__(16) float ptile[8][16 * 36];

  const int lane = threadIdx.x & 31;
  const int wid  = threadIdx.x >> 5;
  const int gid  = blockIdx.x * 8 + wid;       // 0 .. 4095
  const int bh   = gid >> 7;                   // 0 .. 31  (b*H + h)
  const int qb   = gid & 127;                  // query block within (b,h)
  const int q0   = qb * 16;
  const int b    = bh >> 4, h = bh & 15;

  const __bf16* qbase = qh + (size_t)bh * Ss * DHd;
  const __bf16* kbase = kh + (size_t)bh * Ss * DHd;
  const __bf16* vbase = vT + (size_t)bh * DHd * Ss;   // [DH, S]
  float* wbase = wout + (size_t)bh * Ss * Ss;

  // Resident Q fragments (16 rows x DH=64 -> two 16x32 A-fragments).
  const v16bf qf0 = load_a_bf16(qbase, DHd, q0, 0, lane);
  const v16bf qf1 = load_a_bf16(qbase, DHd, q0, 32, lane);

  const v8f vzero = {0.f, 0.f, 0.f, 0.f, 0.f, 0.f, 0.f, 0.f};

  // ---- Pass 1: lane-local online softmax statistics ----
  float lmax[8], lsum[8];
#pragma unroll
  for (int i = 0; i < 8; ++i) { lmax[i] = -__builtin_inff(); lsum[i] = 0.f; }

  constexpr int NB = Ss / 16;  // 128 key blocks
  for (int nb = 0; nb < NB; ++nb) {
    v16bf kf0 = load_b_bf16(kbase, DHd, nb * 16, 0, lane);
    v16bf kf1 = load_b_bf16(kbase, DHd, nb * 16, 32, lane);
    if (nb + 8 < NB) prefetch_b(kbase, DHd, (nb + 8) * 16, 0, lane);
    v8f c = vzero;
    c = wmma_bf16(qf0, kf0, c);
    c = wmma_bf16(qf1, kf1, c);
#pragma unroll
    for (int i = 0; i < 8; ++i) {
      const float x = c[i];
      const float mnew = fmaxf(lmax[i], x);
      lsum[i] = lsum[i] * __expf(lmax[i] - mnew) + __expf(x - mnew);
      lmax[i] = mnew;
    }
  }

  // One cross-lane reduction per row (row lives in one 16-lane group).
  float shiftv[8];
#pragma unroll
  for (int i = 0; i < 8; ++i) {
    float m = lmax[i];
    m = fmaxf(m, __shfl_xor(m, 1));
    m = fmaxf(m, __shfl_xor(m, 2));
    m = fmaxf(m, __shfl_xor(m, 4));
    m = fmaxf(m, __shfl_xor(m, 8));
    float s = lsum[i] * __expf(lmax[i] - m);
    s += __shfl_xor(s, 1);
    s += __shfl_xor(s, 2);
    s += __shfl_xor(s, 4);
    s += __shfl_xor(s, 8);
    shiftv[i] = m + __logf(s);   // p = exp(x - shift) == exp(x-m)/sum
  }

  // ---- Pass 2: normalized weights out + P@V ----
  v8f oacc[4];
#pragma unroll
  for (int j = 0; j < 4; ++j) oacc[j] = vzero;

  float* tile = ptile[wid];  // wave-private 16x32 fp32 tile, row stride 36

  for (int kb2 = 0; kb2 < Ss / 32; ++kb2) {
#pragma unroll
    for (int hv = 0; hv < 2; ++hv) {
      const int nb = kb2 * 2 + hv;
      v16bf kf0 = load_b_bf16(kbase, DHd, nb * 16, 0, lane);
      v16bf kf1 = load_b_bf16(kbase, DHd, nb * 16, 32, lane);
      v8f c = vzero;
      c = wmma_bf16(qf0, kf0, c);
      c = wmma_bf16(qf1, kf1, c);
#pragma unroll
      for (int i = 0; i < 8; ++i) {
        const int row = i + ((lane & 16) ? 8 : 0);
        tile[row * 36 + hv * 16 + (lane & 15)] = __expf(c[i] - shiftv[i]);
      }
    }
    if (kb2 + 4 < Ss / 32) {
      prefetch_b(kbase, DHd, (kb2 + 4) * 32, 0, lane);           // K stream
      prefetch_b(vbase, Ss, 0, (kb2 + 4) * 32, lane);            // V stream
    }

    // Coalesced fp32 weight stores: each 8-lane group writes 128B of one row.
    // (LDS ops are in-order within a wave on CDNA5 -> no barrier needed.)
#pragma unroll
    for (int i = 0; i < 4; ++i) {
      const int flat = i * 128 + lane * 4;
      const int row = flat >> 5, col = flat & 31;
      v4f vv = *(const v4f*)&tile[row * 36 + col];       // ds_load_b128
      *(v4f*)(wbase + (size_t)(q0 + row) * Ss + kb2 * 32 + col) = vv;
    }

    // Re-read tile as A-fragment (16x32) with bf16 conversion.
    const float* pp = &tile[(lane & 15) * 36 + ((lane & 16) ? 16 : 0)];
    v4f y0 = *(const v4f*)pp, y1 = *(const v4f*)(pp + 4);
    v4f y2 = *(const v4f*)(pp + 8), y3 = *(const v4f*)(pp + 12);
    v16bf pf;
#pragma unroll
    for (int i = 0; i < 4; ++i) {
      pf[i] = (__bf16)y0[i]; pf[4 + i] = (__bf16)y1[i];
      pf[8 + i] = (__bf16)y2[i]; pf[12 + i] = (__bf16)y3[i];
    }

    // P (16x32) @ V (32x64): V^T rows give contiguous K-major B-fragments.
#pragma unroll
    for (int j = 0; j < 4; ++j) {
      v16bf vf = load_b_bf16(vbase, Ss, j * 16, kb2 * 32, lane);
      oacc[j] = wmma_bf16(pf, vf, oacc[j]);
    }
  }

  // Store attention output block -> attnX[b][s][h*64+dh] (bf16).
#pragma unroll
  for (int j = 0; j < 4; ++j)
#pragma unroll
    for (int i = 0; i < 8; ++i) {
      const int s  = q0 + i + ((lane & 16) ? 8 : 0);
      const int dh = j * 16 + (lane & 15);
      attnX[((size_t)(b * Ss + s)) * Ee + h * DHd + dh] = (__bf16)oacc[j][i];
    }
}

// ---------------------------------------------------------------------------
// Kernel 3: out = attnX[4096,1024](bf16) @ Wo^T + bo  -> fp32 d_out.
// ---------------------------------------------------------------------------
__global__ void __launch_bounds__(256)
outproj_kernel(const __bf16* __restrict__ X, const float* __restrict__ W,
               const float* __restrict__ bias, float* __restrict__ out) {
  const int lane = threadIdx.x & 31;
  const int m0 = (blockIdx.y * 8 + (threadIdx.x >> 5)) * 32;
  const int n0 = blockIdx.x * 64;

  const v8f vzero = {0.f, 0.f, 0.f, 0.f, 0.f, 0.f, 0.f, 0.f};
  v8f acc[2][4];
#pragma unroll
  for (int mi = 0; mi < 2; ++mi)
#pragma unroll
    for (int j = 0; j < 4; ++j) acc[mi][j] = vzero;

  for (int k0 = 0; k0 < Ee; k0 += 32) {
    v16bf a0 = load_a_bf16(X, Ee, m0, k0, lane);
    v16bf a1 = load_a_bf16(X, Ee, m0 + 16, k0, lane);
    if (k0 + 128 < Ee) {
      __builtin_prefetch((const void*)(X + (size_t)(m0 + (lane & 15)) * Ee + k0 + 128), 0, 1);
      __builtin_prefetch((const void*)(W + (size_t)(n0 + (lane & 15)) * Ee + k0 + 128), 0, 1);
    }
#pragma unroll
    for (int j = 0; j < 4; ++j) {
      v16bf bf = load_b_f32(W, Ee, n0 + j * 16, k0, lane);
      acc[0][j] = wmma_bf16(a0, bf, acc[0][j]);
      acc[1][j] = wmma_bf16(a1, bf, acc[1][j]);
    }
  }

#pragma unroll
  for (int j = 0; j < 4; ++j) {
    const int n = n0 + j * 16 + (lane & 15);
    const float bn = bias[n];
#pragma unroll
    for (int mi = 0; mi < 2; ++mi)
#pragma unroll
      for (int i = 0; i < 8; ++i) {
        const int m = m0 + mi * 16 + i + ((lane & 16) ? 8 : 0);
        out[(size_t)m * Ee + n] = acc[mi][j][i] + bn;
      }
  }
}

// ---------------------------------------------------------------------------
extern "C" void kernel_launch(void* const* d_in, const int* in_sizes, int n_in,
                              void* d_out, int out_size, void* d_ws, size_t ws_size,
                              hipStream_t stream) {
  const float* q  = (const float*)d_in[0];
  const float* k  = (const float*)d_in[1];
  const float* v  = (const float*)d_in[2];
  const float* Wq = (const float*)d_in[3];
  const float* bq = (const float*)d_in[4];
  const float* Wk = (const float*)d_in[5];
  const float* bk = (const float*)d_in[6];
  const float* Wv = (const float*)d_in[7];
  const float* bv = (const float*)d_in[8];
  const float* Wo = (const float*)d_in[9];
  const float* bo = (const float*)d_in[10];

  float* out     = (float*)d_out;                       // [B,S,E]
  float* weights = out + (size_t)Bb * Ss * Ee;          // [B,H,S,S]

  const size_t headElems = (size_t)Bb * Hh * Ss * DHd;  // 4M elems
  __bf16* qh = (__bf16*)d_ws;                           // [B,H,S,DH]
  __bf16* kh = qh + headElems;                          // [B,H,S,DH]
  __bf16* vT = kh + headElems;                          // [B,H,DH,S]
  __bf16* aX = vT + headElems;                          // [B,S,E]
  (void)in_sizes; (void)n_in; (void)out_size; (void)ws_size;  // needs 32MB ws

  const dim3 gproj(Ee / 64, (Bb * Ss) / (32 * 8));      // (16, 16)
  const float scale = 0.125f;                           // DH^-0.5, DH=64

  proj_kernel<<<gproj, 256, 0, stream>>>(q, Wq, bq, qh, scale, 0);
  proj_kernel<<<gproj, 256, 0, stream>>>(k, Wk, bk, kh, 1.0f, 0);
  proj_kernel<<<gproj, 256, 0, stream>>>(v, Wv, bv, vT, 1.0f, 1);

  attn_kernel<<<(Bb * Hh * (Ss / 16)) / 8, 256, 0, stream>>>(qh, kh, vT, weights, aX);

  outproj_kernel<<<gproj, 256, 0, stream>>>(aX, Wo, bo, out);
}
